// TTWindowAttention_63737314673311
// MI455X (gfx1250) — compile-verified, matrix operand order checked by reading, not run on hardware
//
#include <hip/hip_runtime.h>
#include <stdint.h>

// ---------------------------------------------------------------------------
// Swin window attention for MI455X (gfx1250, wave32, WMMA bf16 16x16x32)
// ---------------------------------------------------------------------------

typedef __attribute__((ext_vector_type(16))) __bf16 v16bf;
typedef __attribute__((ext_vector_type(2)))  __bf16 v2bf;
typedef __attribute__((ext_vector_type(8)))  float  v8f;

#define DIM    512
#define NTOK   49
#define NH     16
#define HD     32
#define NWIN   4096
#define MTOT   (NWIN * NTOK)      // 200704 tokens
#define QSCALE 0.17677669529663687f   // 32^-0.5

// ---- helpers ---------------------------------------------------------------

// Native float->bf16 convert; clang lowers to v_cvt_*bf16_f32 on gfx1250.
__device__ __forceinline__ unsigned short f2bf(float f) {
    union { __bf16 h; unsigned short u; } x;
    x.h = (__bf16)f;
    return x.u;
}
// Packed pair convert: builds <2 x bf16> so clang emits one v_cvt_pk_bf16_f32
// producing the packed 32-bit word directly (no scalar repack ALU ops).
__device__ __forceinline__ unsigned int pk2(float a, float b) {
    union { v2bf v; unsigned int u; } x;
    x.v[0] = (__bf16)a;
    x.v[1] = (__bf16)b;
    return x.u;
}
__device__ __forceinline__ v8f zero8() {
    v8f z;
    #pragma unroll
    for (int q = 0; q < 8; ++q) z[q] = 0.f;
    return z;
}

// Load a 16x32 bf16 A/B fragment from LDS (row-major, `stride` in ushorts).
// ISA layout: lane L holds row (L&15); K chunks [g*8, g*8+8) and
// [16+g*8, 16+g*8+8) with g = L>>4, packed as K-pairs into 8 VGPRs.
__device__ __forceinline__ v16bf ld_frag(const unsigned short* p, int stride) {
    const int lane = threadIdx.x & 31;
    const int r = lane & 15, g = lane >> 4;
    const unsigned short* q = p + r * stride + g * 8;
    union { v16bf f; uint4 u[2]; } x;
    x.u[0] = *(const uint4*)(q);
    x.u[1] = *(const uint4*)(q + 16);
    return x.f;
}

__device__ __forceinline__ v8f wmma_bf16(v16bf a, v16bf b, v8f c) {
    // (neg_a, A, neg_b, B, c_mod, C, reuse_a, reuse_b)
    return __builtin_amdgcn_wmma_f32_16x16x32_bf16(false, a, false, b,
                                                   (short)0, c, false, false);
}

// ---- kernel 1: weights -> transposed bf16 ----------------------------------
// Wqt[n*512+k] = bf16(qkv_w[k*1536+n]);  Wpt[n*512+k] = bf16(proj_w[k*512+n])

__global__ void __launch_bounds__(256) convert_weights_kernel(
    const float* __restrict__ qkv_w, const float* __restrict__ proj_w,
    unsigned short* __restrict__ Wqt, unsigned short* __restrict__ Wpt)
{
    int i = blockIdx.x * 256 + threadIdx.x;
    const int NQ = 3 * DIM * DIM;                 // 786432
    if (i < NQ) {
        int n = i / DIM, k = i - n * DIM;
        Wqt[i] = f2bf(qkv_w[(size_t)k * (3 * DIM) + n]);
    } else {
        int j = i - NQ;
        if (j < DIM * DIM) {
            int n = j / DIM, k = j - n * DIM;
            Wpt[j] = f2bf(proj_w[(size_t)k * DIM + n]);
        }
    }
}

// ---- kernel 2: fused QKV projection GEMM -----------------------------------
// X (200704x512 f32) @ Wqt^T (512x1536) + bias -> Q/K/V bf16, per-(win,head)
// layout (NWIN, NH, NTOK, HD); Q pre-scaled by 1/sqrt(HD).
// Block tile 128x128, BK=32, 8 waves arranged 4(m) x 2(n); each wave owns a
// 32x64 strip = 2 A-frags x 4 B-frags -> 8 WMMAs per K-step.

__global__ void __launch_bounds__(256) qkv_gemm_kernel(
    const float* __restrict__ X, const unsigned short* __restrict__ Wt,
    const float* __restrict__ bias,
    unsigned short* __restrict__ Qb, unsigned short* __restrict__ Kb,
    unsigned short* __restrict__ Vb)
{
    __shared__ __align__(16) unsigned short As[128 * 48];
    __shared__ __align__(16) unsigned short Bs[128 * 48];

    const int tid  = threadIdx.x;
    const int m0   = blockIdx.x * 128;
    const int n0   = blockIdx.y * 128;
    const int wave = tid >> 5, lane = tid & 31;
    const int wm   = wave & 3, wn = wave >> 2;
    const int g    = lane >> 4, ln = lane & 15;

    v8f acc[2][4];
    #pragma unroll
    for (int s = 0; s < 2; ++s)
        #pragma unroll
        for (int j = 0; j < 4; ++j) acc[s][j] = zero8();

    for (int kk = 0; kk < DIM; kk += 32) {
        // stage A: 128 rows x 32 K, fp32 -> bf16 (512 chunks of 8, 2/thread)
        #pragma unroll
        for (int c = tid; c < 512; c += 256) {
            int r = c >> 2, kc = (c & 3) * 8;
            const float4* s = (const float4*)(X + (size_t)(m0 + r) * DIM + kk + kc);
            float4 f0 = s[0], f1 = s[1];
            uint4 p;
            p.x = pk2(f0.x, f0.y); p.y = pk2(f0.z, f0.w);
            p.z = pk2(f1.x, f1.y); p.w = pk2(f1.z, f1.w);
            *(uint4*)&As[r * 48 + kc] = p;
        }
        // stage B: 128 rows x 32 K (already bf16; W^T rows contiguous in K)
        #pragma unroll
        for (int c = tid; c < 512; c += 256) {
            int br = c >> 2, bk = (c & 3) * 8;
            *(uint4*)&Bs[br * 48 + bk] =
                *(const uint4*)(Wt + (size_t)(n0 + br) * DIM + kk + bk);
        }
        if (kk + 32 < DIM) {   // -> global_prefetch_b8 for next K slab
            __builtin_prefetch(X  + (size_t)(m0 + (tid >> 1)) * DIM + kk + 32, 0, 1);
            __builtin_prefetch(Wt + (size_t)(n0 + (tid >> 1)) * DIM + kk + 32, 0, 1);
        }
        __syncthreads();

        v16bf a0 = ld_frag(As + (wm * 32 +  0) * 48, 48);
        v16bf a1 = ld_frag(As + (wm * 32 + 16) * 48, 48);
        #pragma unroll
        for (int j = 0; j < 4; ++j) {
            v16bf b = ld_frag(Bs + (wn * 64 + j * 16) * 48, 48);
            acc[0][j] = wmma_bf16(a0, b, acc[0][j]);
            acc[1][j] = wmma_bf16(a1, b, acc[1][j]);
        }
        __syncthreads();
    }

    // epilogue: scatter into Q/K/V (NWIN, NH, NTOK, HD) bf16
    #pragma unroll
    for (int s = 0; s < 2; ++s) {
        #pragma unroll
        for (int j = 0; j < 4; ++j) {
            #pragma unroll
            for (int v = 0; v < 8; ++v) {
                int m = m0 + wm * 32 + s * 16 + v + 8 * g;
                int n = n0 + wn * 64 + j * 16 + ln;
                float val = acc[s][j][v] + bias[n];
                int trip = n >> 9;               // 0=q 1=k 2=v
                int h = (n >> 5) & 15, d = n & 31;
                if (trip == 0) val *= QSCALE;
                int w = m / NTOK, tok = m - w * NTOK;
                unsigned short* dst = (trip == 0) ? Qb : ((trip == 1) ? Kb : Vb);
                dst[(((size_t)w * NH + h) * NTOK + tok) * HD + d] = f2bf(val);
            }
        }
    }
}

// ---- kernel 3: windowed attention ------------------------------------------
// One block per (window, head). 49 tokens padded to 64; zero-padded rows make
// the pad lanes inert in both WMMA GEMMs; softmax only runs over real columns.

__global__ void __launch_bounds__(256) attn_kernel(
    const unsigned short* __restrict__ Qb, const unsigned short* __restrict__ Kb,
    const unsigned short* __restrict__ Vb, const float* __restrict__ rbias,
    const float* __restrict__ mask, unsigned short* __restrict__ AOb)
{
    __shared__ __align__(16) unsigned short Qs[64 * 48];
    __shared__ __align__(16) unsigned short Ks[64 * 48];
    __shared__ __align__(16) unsigned short Vt[32 * 72];   // V transposed (hd, tok)
    __shared__ __align__(16) float          Sf[64 * 68];
    __shared__ __align__(16) unsigned short Ps[64 * 72];   // softmax probs bf16

    const int w = blockIdx.x, h = blockIdx.y;
    const int tid  = threadIdx.x;
    const int wave = tid >> 5, lane = tid & 31;
    const int g    = lane >> 4, ln = lane & 15;
    const size_t base = ((size_t)w * NH + h) * NTOK * HD;

    // phase 0: zero pad regions (probs buffer, V^T, Q/K pad rows)
    for (int i = tid; i < 64 * 72 / 2; i += 256) ((unsigned int*)Ps)[i] = 0;
    for (int i = tid; i < 32 * 72 / 2; i += 256) ((unsigned int*)Vt)[i] = 0;
    for (int i = tid; i < 15 * 48 / 2; i += 256) {
        ((unsigned int*)(Qs + 49 * 48))[i] = 0;
        ((unsigned int*)(Ks + 49 * 48))[i] = 0;
    }
    __syncthreads();

    // phase 1: stage Q, K (row-major) and V transposed
    if (tid < 196) {                     // 49 rows x 4 chunks of 8 bf16
        int tok = tid >> 2, kc = (tid & 3) * 8;
        *(uint4*)&Qs[tok * 48 + kc] = *(const uint4*)(Qb + base + tok * HD + kc);
        *(uint4*)&Ks[tok * 48 + kc] = *(const uint4*)(Kb + base + tok * HD + kc);
    }
    for (int i = tid; i < NTOK * HD; i += 256) {
        int tok = i >> 5, d = i & 31;
        Vt[d * 72 + tok] = Vb[base + i];
    }
    __syncthreads();

    // phase 2: S = Q @ K^T   (64x64 from 4x4 tiles; 2 tiles per wave)
    #pragma unroll
    for (int t = 0; t < 2; ++t) {
        int idx = wave * 2 + t;
        int tm = idx >> 2, tn = idx & 3;
        v16bf a = ld_frag(Qs + tm * 16 * 48, 48);
        v16bf b = ld_frag(Ks + tn * 16 * 48, 48);
        v8f s = wmma_bf16(a, b, zero8());
        #pragma unroll
        for (int v = 0; v < 8; ++v)
            Sf[(tm * 16 + v + 8 * g) * 68 + tn * 16 + ln] = s[v];
    }
    __syncthreads();

    // phase 3: bias + shift-mask + softmax (fp32), probs -> bf16
    if (tid < NTOK) {
        float* srow = Sf + tid * 68;
        const float* br = rbias + ((size_t)h * NTOK + tid) * NTOK;
        const float* mr = mask  + ((size_t)(w & 63) * NTOK + tid) * NTOK;
        float mx = -1e30f;
        for (int c = 0; c < NTOK; ++c) {
            float x = srow[c] + br[c] + mr[c];
            srow[c] = x;
            mx = fmaxf(mx, x);
        }
        float sum = 0.f;
        for (int c = 0; c < NTOK; ++c) sum += __expf(srow[c] - mx);
        float rs = 1.f / sum;
        for (int c = 0; c < NTOK; ++c)
            Ps[tid * 72 + c] = f2bf(__expf(srow[c] - mx) * rs);
    }
    __syncthreads();

    // phase 4: O = P @ V   (64x32 from 4x2 tiles; 1 tile per wave; K = 64)
    {
        int tm = wave >> 1, tn = wave & 1;
        v8f o = zero8();
        #pragma unroll
        for (int ks = 0; ks < 2; ++ks) {
            v16bf a = ld_frag(Ps + tm * 16 * 72 + ks * 32, 72);
            v16bf b = ld_frag(Vt + tn * 16 * 72 + ks * 32, 72);
            o = wmma_bf16(a, b, o);
        }
        #pragma unroll
        for (int v = 0; v < 8; ++v) {
            int tok = tm * 16 + v + 8 * g;
            if (tok < NTOK)
                AOb[((size_t)w * NTOK + tok) * DIM + h * HD + tn * 16 + ln] =
                    f2bf(o[v]);
        }
    }
}

// ---- kernel 4: output projection GEMM --------------------------------------
// AO (200704x512 bf16) @ Wpt^T (512x512) + bias -> d_out fp32
// Same 128x128 tiling as the QKV GEMM; A is already bf16.

__global__ void __launch_bounds__(256) proj_gemm_kernel(
    const unsigned short* __restrict__ A, const unsigned short* __restrict__ Wt,
    const float* __restrict__ bias, float* __restrict__ out)
{
    __shared__ __align__(16) unsigned short As[128 * 48];
    __shared__ __align__(16) unsigned short Bs[128 * 48];

    const int tid  = threadIdx.x;
    const int m0   = blockIdx.x * 128;
    const int n0   = blockIdx.y * 128;
    const int wave = tid >> 5, lane = tid & 31;
    const int wm   = wave & 3, wn = wave >> 2;
    const int g    = lane >> 4, ln = lane & 15;

    v8f acc[2][4];
    #pragma unroll
    for (int s = 0; s < 2; ++s)
        #pragma unroll
        for (int j = 0; j < 4; ++j) acc[s][j] = zero8();

    for (int kk = 0; kk < DIM; kk += 32) {
        #pragma unroll
        for (int c = tid; c < 512; c += 256) {
            int r = c >> 2, kc = (c & 3) * 8;
            *(uint4*)&As[r * 48 + kc] =
                *(const uint4*)(A + (size_t)(m0 + r) * DIM + kk + kc);
        }
        #pragma unroll
        for (int c = tid; c < 512; c += 256) {
            int br = c >> 2, bk = (c & 3) * 8;
            *(uint4*)&Bs[br * 48 + bk] =
                *(const uint4*)(Wt + (size_t)(n0 + br) * DIM + kk + bk);
        }
        if (kk + 32 < DIM) {
            __builtin_prefetch(A  + (size_t)(m0 + (tid >> 1)) * DIM + kk + 32, 0, 1);
            __builtin_prefetch(Wt + (size_t)(n0 + (tid >> 1)) * DIM + kk + 32, 0, 1);
        }
        __syncthreads();

        v16bf a0 = ld_frag(As + (wm * 32 +  0) * 48, 48);
        v16bf a1 = ld_frag(As + (wm * 32 + 16) * 48, 48);
        #pragma unroll
        for (int j = 0; j < 4; ++j) {
            v16bf b = ld_frag(Bs + (wn * 64 + j * 16) * 48, 48);
            acc[0][j] = wmma_bf16(a0, b, acc[0][j]);
            acc[1][j] = wmma_bf16(a1, b, acc[1][j]);
        }
        __syncthreads();
    }

    #pragma unroll
    for (int s = 0; s < 2; ++s) {
        #pragma unroll
        for (int j = 0; j < 4; ++j) {
            #pragma unroll
            for (int v = 0; v < 8; ++v) {
                int m = m0 + wm * 32 + s * 16 + v + 8 * g;
                int n = n0 + wn * 64 + j * 16 + ln;
                out[(size_t)m * DIM + n] = acc[s][j][v] + bias[n];
            }
        }
    }
}

// ---- launcher --------------------------------------------------------------

extern "C" void kernel_launch(void* const* d_in, const int* in_sizes, int n_in,
                              void* d_out, int out_size, void* d_ws, size_t ws_size,
                              hipStream_t stream)
{
    const float* X      = (const float*)d_in[0];
    const float* mask   = (const float*)d_in[1];
    const float* qkv_w  = (const float*)d_in[2];
    const float* qkv_b  = (const float*)d_in[3];
    const float* proj_w = (const float*)d_in[4];
    const float* proj_b = (const float*)d_in[5];
    const float* rbias  = (const float*)d_in[6];
    float* out = (float*)d_out;

    // workspace layout (bf16 / ushort elements)
    unsigned short* Wqt = (unsigned short*)d_ws;            // 1536*512
    unsigned short* Wpt = Wqt + (size_t)3 * DIM * DIM;      // 512*512
    unsigned short* Qb  = Wpt + (size_t)DIM * DIM;          // MTOT*512 each
    unsigned short* Kb  = Qb  + (size_t)MTOT * DIM;
    unsigned short* Vb  = Kb  + (size_t)MTOT * DIM;
    unsigned short* AOb = Vb  + (size_t)MTOT * DIM;

    convert_weights_kernel<<<4096, 256, 0, stream>>>(qkv_w, proj_w, Wqt, Wpt);
    qkv_gemm_kernel<<<dim3(MTOT / 128, (3 * DIM) / 128), 256, 0, stream>>>(
        X, Wqt, qkv_b, Qb, Kb, Vb);
    attn_kernel<<<dim3(NWIN, NH), 256, 0, stream>>>(Qb, Kb, Vb, rbias, mask, AOb);
    proj_gemm_kernel<<<dim3(MTOT / 128, DIM / 128), 256, 0, stream>>>(
        AOb, Wpt, proj_b, out);
}
